// CrossAttention_68238440399178
// MI455X (gfx1250) — matrix operand order repaired and checked by reading, not compiled, not run on hardware
//
#include <hip/hip_runtime.h>
#include <hip/hip_bf16.h>

// Problem constants (reference: B,SQ,SKV,E,H = 4,2048,2048,1024,16; HD=64)
#define BB   4
#define SQL  2048
#define SKVL 2048
#define EE   1024
#define HH   16
#define HDD  64

typedef __attribute__((ext_vector_type(16))) __bf16 v16bf;
typedef __attribute__((ext_vector_type(8)))  __bf16 v8bf;
typedef __attribute__((ext_vector_type(8)))  float  v8f;

__device__ __forceinline__ v8f wmma_bf16(v16bf a, v16bf b, v8f c) {
  // (neg_a, A, neg_b, B, c_mod, C, reuse_a, reuse_b)
  return __builtin_amdgcn_wmma_f32_16x16x32_bf16(false, a, false, b, (short)0, c,
                                                 false, false);
}

__device__ __forceinline__ v8f zero8() {
  v8f z;
#pragma unroll
  for (int i = 0; i < 8; ++i) z[i] = 0.0f;
  return z;
}

// ---- A fragment: 16x32 (MxK) bf16. ISA layout:
// lane<16 : row M=lane,    K = {0..7, 16..23}
// lane>=16: row M=lane-16, K = {8..15, 24..31}
__device__ __forceinline__ v16bf a_frag(const __bf16* p, int ld) {
  const int lane = threadIdx.x & 31;
  const __bf16* row = p + (size_t)(lane & 15) * ld + ((lane & 16) ? 8 : 0);
  v16bf a;
#pragma unroll
  for (int j = 0; j < 8; ++j) {
    a[j]     = row[j];
    a[j + 8] = row[j + 16];
  }
  return a;
}

// ---- B fragment: 32x16 (KxN) bf16, element (k,n) = p[n*ld + k].
// lane<16 : column N=lane,    K = {0..7, 16..23}
// lane>=16: column N=lane-16, K = {8..15, 24..31}
__device__ __forceinline__ v16bf b_frag(const __bf16* p, int ld) {
  const int lane = threadIdx.x & 31;
  const __bf16* row = p + (size_t)(lane & 15) * ld + ((lane & 16) ? 8 : 0);
  v16bf b;
#pragma unroll
  for (int j = 0; j < 8; ++j) {
    b[j]     = row[j];
    b[j + 8] = row[j + 16];
  }
  return b;
}

// =====================================================================
// Elementwise fp32 -> bf16 conversion, 8 elements/thread, b128 in/out.
// =====================================================================
__global__ __launch_bounds__(256) void cvt_kernel(const float* __restrict__ in,
                                                  __bf16* __restrict__ out,
                                                  long n) {
  const long i = ((long)blockIdx.x * blockDim.x + threadIdx.x) * 8;
  if (i >= n) return;
  const float4 a = ((const float4*)(in + i))[0];
  const float4 b = ((const float4*)(in + i))[1];
  v8bf o;
  o[0] = (__bf16)a.x; o[1] = (__bf16)a.y; o[2] = (__bf16)a.z; o[3] = (__bf16)a.w;
  o[4] = (__bf16)b.x; o[5] = (__bf16)b.y; o[6] = (__bf16)b.z; o[7] = (__bf16)b.w;
  *(v8bf*)(out + i) = o;
}

// =====================================================================
// Projection GEMM (all-bf16 operands): Y[m,n] = sum_k X[m,k]*W[n,k] + bias[n]
// One wave computes a 32(M) x 64(N) tile.  K in steps of 32.
// OUT_MODE: 0 = bf16 row-major, 1 = bf16 transposed per-batch [N][Srows],
//           2 = fp32 row-major
// =====================================================================
template <int OUT_MODE>
__global__ __launch_bounds__(256) void proj_kernel(
    const __bf16* __restrict__ X, const __bf16* __restrict__ W,
    const float* __restrict__ bias, void* __restrict__ Yv,
    int M, int N, int K, int Srows) {
  const int wid = (blockIdx.x * blockDim.x + threadIdx.x) >> 5;
  const int tiles_n = N >> 6;
  const int mt = wid / tiles_n;
  const int nt = wid % tiles_n;
  if (mt * 32 >= M) return;

  const int lane  = threadIdx.x & 31;
  const int nlo   = lane & 15;
  const int mhalf = (lane >> 4) & 1;

  v8f acc[2][4];
#pragma unroll
  for (int i = 0; i < 2; ++i)
#pragma unroll
    for (int t = 0; t < 4; ++t) acc[i][t] = zero8();

  const __bf16* x0 = X + (size_t)mt * 32 * K;
  const __bf16* w0 = W + (size_t)nt * 64 * K;

  for (int k0 = 0; k0 < K; k0 += 32) {
    v16bf a0 = a_frag(x0 + k0, K);
    v16bf a1 = a_frag(x0 + (size_t)16 * K + k0, K);
#pragma unroll
    for (int t = 0; t < 4; ++t) {
      v16bf bt = b_frag(w0 + (size_t)t * 16 * K + k0, K);
      acc[0][t] = wmma_bf16(a0, bt, acc[0][t]);
      acc[1][t] = wmma_bf16(a1, bt, acc[1][t]);
    }
  }

#pragma unroll
  for (int t = 0; t < 4; ++t) {
    const float bv = bias[nt * 64 + t * 16 + nlo];
#pragma unroll
    for (int i = 0; i < 2; ++i)
#pragma unroll
      for (int r = 0; r < 8; ++r) {
        const int m = mt * 32 + i * 16 + mhalf * 8 + r;
        const int n = nt * 64 + t * 16 + nlo;
        const float v = acc[i][t][r] + bv;
        if (OUT_MODE == 0) {
          ((__bf16*)Yv)[(size_t)m * N + n] = (__bf16)v;
        } else if (OUT_MODE == 1) {
          const int bb = m / Srows, mr = m % Srows;
          ((__bf16*)Yv)[((size_t)bb * N + n) * Srows + mr] = (__bf16)v;
        } else {
          ((float*)Yv)[(size_t)m * N + n] = v;
        }
      }
  }
}

// =====================================================================
// StableMax attention, one-pass streaming (no softmax rescale needed):
//   num(x) = x>=0 ? x+1 : 1/(1-x)   (x = score/8)   -- branchless cndmask
//   ctx = (sum_kv num * V) / (sum_kv num)
// One wave owns a 16-row Q tile for one (b,h); streams KV in 32-chunks.
// =====================================================================
__global__ __launch_bounds__(256) void attn_kernel(
    const __bf16* __restrict__ Q, const __bf16* __restrict__ Kb,
    const __bf16* __restrict__ VT, __bf16* __restrict__ CTX) {
  __shared__ __bf16 lds[8][16 * 40];  // per-wave padded 16x32 P tile
  const int wslot = (threadIdx.x >> 5) & 7;
  const int wid   = (blockIdx.x * blockDim.x + threadIdx.x) >> 5;
  const int qtiles = SQL / 16;  // 128
  const int qt = wid % qtiles;
  const int h  = (wid / qtiles) % HH;
  const int b  = wid / (qtiles * HH);

  const int lane  = threadIdx.x & 31;
  const int nlo   = lane & 15;
  const int mhalf = (lane >> 4) & 1;

  // Q fragments are loop-invariant over KV (K = HD = 64 -> two 32-steps)
  const __bf16* qp = Q + ((size_t)b * SQL + (size_t)qt * 16) * EE + h * HDD;
  const v16bf aq0 = a_frag(qp, EE);
  const v16bf aq1 = a_frag(qp + 32, EE);

  const __bf16* kp = Kb + (size_t)b * SKVL * EE + h * HDD;
  const __bf16* vp = VT + ((size_t)b * EE + h * HDD) * SKVL;

  v8f ctx[4];
  float rs[8];
#pragma unroll
  for (int t = 0; t < 4; ++t) ctx[t] = zero8();
#pragma unroll
  for (int r = 0; r < 8; ++r) rs[r] = 0.0f;

  __bf16* my = lds[wslot];

  for (int kv0 = 0; kv0 < SKVL; kv0 += 32) {
    // ---- scores: S(16 x 32) = Q(16x64) . K^T(64x32)
    v8f S[2];
#pragma unroll
    for (int t = 0; t < 2; ++t) {
      const __bf16* kt = kp + (size_t)(kv0 + t * 16) * EE;
      v16bf bk0 = b_frag(kt, EE);       // d = 0..31
      v16bf bk1 = b_frag(kt + 32, EE);  // d = 32..63
      v8f s = zero8();
      s = wmma_bf16(aq0, bk0, s);
      s = wmma_bf16(aq1, bk1, s);
      S[t] = s;
    }

    // ---- stablemax numerator (branchless), row-sum accum, P -> LDS (bf16)
#pragma unroll
    for (int t = 0; t < 2; ++t)
#pragma unroll
      for (int r = 0; r < 8; ++r) {
        const float x  = S[t][r] * 0.125f;  // 1/sqrt(64)
        const float pa = x + 1.0f;
        const float pb = __builtin_amdgcn_rcpf(1.0f - x);  // v_rcp_f32
        const float p  = (x >= 0.0f) ? pa : pb;            // v_cndmask
        rs[r] += p;
        my[(mhalf * 8 + r) * 40 + t * 16 + nlo] = (__bf16)p;
      }
    asm volatile("s_wait_dscnt 0" ::: "memory");

    // ---- reload P as A fragment (16x32, K = kv-local)
    v16bf ap;
    {
      const __bf16* row = my + (size_t)(lane & 15) * 40 + ((lane & 16) ? 8 : 0);
#pragma unroll
      for (int j = 0; j < 8; ++j) {
        ap[j]     = row[j];
        ap[j + 8] = row[j + 16];
      }
    }
    asm volatile("s_wait_dscnt 0" ::: "memory");

    // ---- ctx += P(16x32) . V(32x64)   (V stored transposed: VT[d][kv])
#pragma unroll
    for (int dt = 0; dt < 4; ++dt) {
      v16bf bv = b_frag(vp + (size_t)dt * 16 * SKVL + kv0, SKVL);
      ctx[dt] = wmma_bf16(ap, bv, ctx[dt]);
    }
  }

  // ---- full row sums: reduce across the 16 lanes of each half
#pragma unroll
  for (int r = 0; r < 8; ++r) {
    float v = rs[r];
    v += __shfl_xor(v, 1, 32);
    v += __shfl_xor(v, 2, 32);
    v += __shfl_xor(v, 4, 32);
    v += __shfl_xor(v, 8, 32);
    rs[r] = 1.0f / v;  // numerators are strictly positive
  }

  // ---- normalize + store context (bf16, heads re-interleaved into E)
  __bf16* op = CTX + ((size_t)b * SQL + (size_t)qt * 16) * EE + h * HDD;
#pragma unroll
  for (int dt = 0; dt < 4; ++dt)
#pragma unroll
    for (int r = 0; r < 8; ++r) {
      const int m = mhalf * 8 + r;
      op[(size_t)m * EE + dt * 16 + nlo] = (__bf16)(ctx[dt][r] * rs[r]);
    }
}

extern "C" void kernel_launch(void* const* d_in, const int* in_sizes, int n_in,
                              void* d_out, int out_size, void* d_ws, size_t ws_size,
                              hipStream_t stream) {
  const float* query     = (const float*)d_in[0];
  const float* key_value = (const float*)d_in[1];
  const float* Wq = (const float*)d_in[2];
  const float* bq = (const float*)d_in[3];
  const float* Wk = (const float*)d_in[4];
  const float* bk = (const float*)d_in[5];
  const float* Wv = (const float*)d_in[6];
  const float* bv = (const float*)d_in[7];
  const float* Wo = (const float*)d_in[8];
  const float* bo = (const float*)d_in[9];

  const size_t act = (size_t)BB * SQL * EE;   // 8M elements
  const size_t wsz = (size_t)EE * EE;         // 1M elements
  const size_t need = (4 * act + 2 * act + 4 * wsz) * sizeof(__bf16);  // ~104 MB
  if (ws_size < need) return;

  __bf16* Qbf  = (__bf16*)d_ws;
  __bf16* Kbf  = Qbf + act;
  __bf16* VT   = Kbf + act;
  __bf16* CTX  = VT + act;
  __bf16* Xq   = CTX + act;      // bf16 copy of query
  __bf16* Xkv  = Xq + act;       // bf16 copy of key_value
  __bf16* Wqb  = Xkv + act;
  __bf16* Wkb  = Wqb + wsz;
  __bf16* Wvb  = Wkb + wsz;
  __bf16* Wob  = Wvb + wsz;

  const int M = BB * SQL, N = EE, K = EE;
  const dim3 blk(256);
  const int cvt_act_blocks = (int)(act / 8 / 256);  // 4096
  const int cvt_w_blocks   = (int)(wsz / 8 / 256);  // 512
  const int proj_blocks    = (M / 32) * (N / 64) / 8;     // 512
  const int attn_blocks    = (BB * HH * (SQL / 16)) / 8;  // 1024

  // fp32 -> bf16 pre-conversion (done once; removes per-tile cvt VALU)
  cvt_kernel<<<cvt_act_blocks, blk, 0, stream>>>(query,     Xq,  (long)act);
  cvt_kernel<<<cvt_act_blocks, blk, 0, stream>>>(key_value, Xkv, (long)act);
  cvt_kernel<<<cvt_w_blocks,   blk, 0, stream>>>(Wq, Wqb, (long)wsz);
  cvt_kernel<<<cvt_w_blocks,   blk, 0, stream>>>(Wk, Wkb, (long)wsz);
  cvt_kernel<<<cvt_w_blocks,   blk, 0, stream>>>(Wv, Wvb, (long)wsz);
  cvt_kernel<<<cvt_w_blocks,   blk, 0, stream>>>(Wo, Wob, (long)wsz);

  // projections
  proj_kernel<0><<<proj_blocks, blk, 0, stream>>>(Xq,  Wqb, bq, Qbf, M, N, K, SQL);
  proj_kernel<0><<<proj_blocks, blk, 0, stream>>>(Xkv, Wkb, bk, Kbf, M, N, K, SKVL);
  proj_kernel<1><<<proj_blocks, blk, 0, stream>>>(Xkv, Wvb, bv, VT,  M, N, K, SKVL);

  // streaming stablemax attention
  attn_kernel<<<attn_blocks, blk, 0, stream>>>(Qbf, Kbf, VT, CTX);

  // output projection (fp32 out)
  proj_kernel<2><<<proj_blocks, blk, 0, stream>>>(CTX, Wob, bo, d_out, M, N, K, SQL);
}
